// Model_78176994722019
// MI455X (gfx1250) — matrix operand order repaired and checked by reading, not compiled, not run on hardware
//
#include <hip/hip_runtime.h>
#include <math.h>

typedef __bf16 bf16_t;
typedef __attribute__((ext_vector_type(16))) bf16_t v16bf;
typedef __attribute__((ext_vector_type(8)))  bf16_t v8bf;
typedef __attribute__((ext_vector_type(8)))  float  v8f;
typedef __attribute__((ext_vector_type(4)))  unsigned int v4u;
typedef __attribute__((ext_vector_type(8)))  int v8i;
typedef __attribute__((ext_vector_type(4)))  int v4i;

#define DEV __device__ __forceinline__

// ---- bf16 <-> f32 via bit ops (round-to-nearest-even) ----
DEV bf16_t f2bf(float f) {
    unsigned u = __builtin_bit_cast(unsigned, f);
    unsigned r = (u + 0x7FFFu + ((u >> 16) & 1u)) >> 16;
    unsigned short s = (unsigned short)r;
    return __builtin_bit_cast(bf16_t, s);
}
DEV float bf2f(bf16_t b) {
    unsigned short s = __builtin_bit_cast(unsigned short, b);
    unsigned u = ((unsigned)s) << 16;
    return __builtin_bit_cast(float, u);
}

// ---- WMMA fragment loaders (contiguous-K order, same for A and B, wave32) ----
// A fragment 16x32 (MxK): lanes 0-15 -> M=lane, K in [0..15]
//                         lanes 16-31 -> M=lane-16, K in [16..31]
DEV v16bf load_frag_a(const bf16_t* __restrict__ A, int lda, int row0, int k0, int lane) {
    int m  = lane & 15;
    int kh = (lane >> 4) & 1;
    const bf16_t* p = A + (long)(row0 + m) * lda + k0 + kh * 16;
    v8bf lo = *(const v8bf*)(p);
    v8bf hi = *(const v8bf*)(p + 8);
    return __builtin_shufflevector(lo, hi, 0,1,2,3,4,5,6,7,8,9,10,11,12,13,14,15);
}
// B fragment 32x16 (KxN) from weights stored transposed (N x K row-major):
// lane = N (lanes 0-15: K=0..15, lanes 16-31: K=16..31), 16 contiguous K per lane.
DEV v16bf load_frag_b(const bf16_t* __restrict__ Wt, int ldb, int n0, int k0, int lane) {
    int n  = lane & 15;
    int kh = (lane >> 4) & 1;
    const bf16_t* p = Wt + (long)(n0 + n) * ldb + k0 + kh * 16;
    return *(const v16bf*)(p);
}

// Accumulate a 32x32 tile (2x2 WMMA frags) over full K (direct-global B path).
DEV void gemm_tile_32x32(const bf16_t* __restrict__ A, int lda,
                         const bf16_t* __restrict__ Wt, int ldb,
                         int row0, int col0, int K, int lane, v8f acc[2][2]) {
    for (int k0 = 0; k0 < K; k0 += 32) {
        v16bf a0 = load_frag_a(A, lda, row0,      k0, lane);
        v16bf a1 = load_frag_a(A, lda, row0 + 16, k0, lane);
        v16bf b0 = load_frag_b(Wt, ldb, col0,      k0, lane);
        v16bf b1 = load_frag_b(Wt, ldb, col0 + 16, k0, lane);
        acc[0][0] = __builtin_amdgcn_wmma_f32_16x16x32_bf16(false, a0, false, b0, (short)0, acc[0][0], false, false);
        acc[0][1] = __builtin_amdgcn_wmma_f32_16x16x32_bf16(false, a0, false, b1, (short)0, acc[0][1], false, false);
        acc[1][0] = __builtin_amdgcn_wmma_f32_16x16x32_bf16(false, a1, false, b0, (short)0, acc[1][0], false, false);
        acc[1][1] = __builtin_amdgcn_wmma_f32_16x16x32_bf16(false, a1, false, b1, (short)0, acc[1][1], false, false);
    }
}

// ---- kernel: transpose + cast weights f32(rows x cols) -> bf16(cols x rows) ----
__global__ void convert_transpose_kernel(const float* __restrict__ src, bf16_t* __restrict__ dst,
                                         int rows, int cols) {
    long idx = (long)blockIdx.x * blockDim.x + threadIdx.x;
    long n = (long)rows * cols;
    if (idx >= n) return;
    int r = (int)(idx / cols);
    int c = (int)(idx % cols);
    dst[(long)c * rows + r] = f2bf(src[idx]);
}

// ---- kernel: rmsnorm per token + cast to bf16 ----
__global__ __launch_bounds__(256)
void rmsnorm_cast_kernel(const float* __restrict__ x, const float* __restrict__ w,
                         bf16_t* __restrict__ xn, int Dd) {
    int row = blockIdx.x;
    const float* xr = x + (long)row * Dd;
    float ss = 0.f;
    for (int i = threadIdx.x; i < Dd; i += 256) { float v = xr[i]; ss += v * v; }
    __shared__ float sm[256];
    sm[threadIdx.x] = ss;
    __syncthreads();
    for (int off = 128; off > 0; off >>= 1) {
        if (threadIdx.x < off) sm[threadIdx.x] += sm[threadIdx.x + off];
        __syncthreads();
    }
    float norm = rsqrtf(sm[0] / (float)Dd + 1e-8f);
    bf16_t* xo = xn + (long)row * Dd;
    for (int i = threadIdx.x; i < Dd; i += 256)
        xo[i] = f2bf(xr[i] * norm * w[i]);
}

// ---- kernel: uzg = xn @ W_in + b_in, split into u/z/g bf16.
// B (weight) tile is streamed into LDS by the Tensor Data Mover (TDM),
// double-buffered 64(N) x 128(K) bf16 chunks, tracked with TENSORcnt.
__global__ __launch_bounds__(128)
void gemm_in_kernel(const bf16_t* __restrict__ xn, const bf16_t* __restrict__ wt_in,
                    const float* __restrict__ b_in,
                    bf16_t* __restrict__ u_bf, bf16_t* __restrict__ z_bf, bf16_t* __restrict__ g_bf,
                    int K) {
    const int KC = 128;                       // K chunk staged per TDM transfer
    __shared__ __align__(128) bf16_t sB[2][64 * KC];   // 2 x 16KB double buffer

    int lane = threadIdx.x & 31;
    int wave = threadIdx.x >> 5;
    int row0  = blockIdx.x * 64 + (wave >> 1) * 32;
    int colB0 = blockIdx.y * 64;              // block's N origin (rows of Wt)
    int colg0 = colB0 + (wave & 1) * 32;
    // Force a scalar branch so only one wave programs the TDM.
    bool tdm_wave = (__builtin_amdgcn_readfirstlane(wave) == 0);

    // D# group1 (constant): data_size=2B, tensor K x 768 (dim0 = K contiguous),
    // dim0 stride = K, tile = KC x 64, groups 2/3/4 zero (dims unused).
    unsigned td0 = (unsigned)K;
    unsigned td1 = 768u;
    v8i g1 = { (int)0x00010000u,                              // wg_mask=0, data_size=1(2B)
               (int)((td0 & 0xFFFFu) << 16),                  // tensor_dim0[15:0]
               (int)((td0 >> 16) | ((td1 & 0xFFFFu) << 16)),  // dim0[31:16] | dim1[15:0]
               (int)((td1 >> 16) | ((unsigned)KC << 16)),     // dim1[31:16] | tile_dim0
               (int)64u,                                      // tile_dim1=64, tile_dim2=0
               (int)td0,                                      // tensor_dim0_stride[31:0]
               0, 0 };                                        // stride[47:32], dim1_stride
    v4i gz4 = { 0, 0, 0, 0 };
    v8i gz8 = { 0, 0, 0, 0, 0, 0, 0, 0 };

    int nch = K / KC;

    auto issue_tdm = [&](int chunk, int buf) {
        unsigned long long ga =
            (unsigned long long)(size_t)(wt_in + (long)colB0 * K + chunk * KC);
        unsigned lds = (unsigned)(size_t)(void*)&sB[buf][0];
        v4u g0 = { (unsigned)__builtin_amdgcn_readfirstlane(1),           // count=1, valid
                   (unsigned)__builtin_amdgcn_readfirstlane((int)lds),    // lds_addr
                   (unsigned)__builtin_amdgcn_readfirstlane((int)(ga & 0xFFFFFFFFull)),
                   (unsigned)__builtin_amdgcn_readfirstlane(
                       (int)(((ga >> 32) & 0x01FFFFFFull) | (2ull << 30))) }; // addr_hi|type=2
        __builtin_amdgcn_tensor_load_to_lds(g0, g1, gz4, gz4, gz8, 0);
    };

    v8f zero = {};
    v8f acc[2][2] = {{zero, zero}, {zero, zero}};

    if (tdm_wave) {
        issue_tdm(0, 0);
        __builtin_amdgcn_s_wait_tensorcnt(0);
    }
    __syncthreads();

    int n = lane & 15, kh = (lane >> 4) & 1;
    int nloc = (wave & 1) * 32 + n;

    for (int c = 0; c < nch; ++c) {
        int cur = c & 1;
        if (tdm_wave && (c + 1 < nch)) issue_tdm(c + 1, cur ^ 1);  // prefetch next chunk
        const bf16_t* Bt = &sB[cur][0];
        #pragma unroll
        for (int kk = 0; kk < KC; kk += 32) {
            int k0 = c * KC + kk;
            v16bf a0 = load_frag_a(xn, K, row0,      k0, lane);
            v16bf a1 = load_frag_a(xn, K, row0 + 16, k0, lane);
            v16bf b0 = *(const v16bf*)(Bt + (long)nloc        * KC + kk + kh * 16);
            v16bf b1 = *(const v16bf*)(Bt + (long)(nloc + 16) * KC + kk + kh * 16);
            acc[0][0] = __builtin_amdgcn_wmma_f32_16x16x32_bf16(false, a0, false, b0, (short)0, acc[0][0], false, false);
            acc[0][1] = __builtin_amdgcn_wmma_f32_16x16x32_bf16(false, a0, false, b1, (short)0, acc[0][1], false, false);
            acc[1][0] = __builtin_amdgcn_wmma_f32_16x16x32_bf16(false, a1, false, b0, (short)0, acc[1][0], false, false);
            acc[1][1] = __builtin_amdgcn_wmma_f32_16x16x32_bf16(false, a1, false, b1, (short)0, acc[1][1], false, false);
        }
        if (tdm_wave) __builtin_amdgcn_s_wait_tensorcnt(0);
        __syncthreads();
    }

    int mh = (lane >> 4) & 1;
    #pragma unroll
    for (int j = 0; j < 2; ++j) {
        int colg = colg0 + j * 16 + n;
        int seg  = colg >> 8;      // 0=u, 1=z, 2=g
        int cs   = colg & 255;
        bf16_t* dst = (seg == 0) ? u_bf : ((seg == 1) ? z_bf : g_bf);
        float bv = b_in[colg];
        #pragma unroll
        for (int i = 0; i < 2; ++i) {
            int rb = row0 + i * 16 + mh * 8;
            #pragma unroll
            for (int v = 0; v < 8; ++v)
                dst[(long)(rb + v) * 256 + cs] = f2bf(acc[i][j][v] + bv);
        }
    }
}

// ---- kernel: gate GEMM (K=N=256) with fused tanh/sigmoid ----
__global__ __launch_bounds__(128)
void gemm_gate_kernel(const bf16_t* __restrict__ A, const bf16_t* __restrict__ Wt,
                      const float* __restrict__ bias, float* __restrict__ outp, int act) {
    const int K = 256;
    int lane = threadIdx.x & 31;
    int wave = threadIdx.x >> 5;
    int row0 = blockIdx.x * 64 + (wave >> 1) * 32;
    int col0 = blockIdx.y * 64 + (wave & 1) * 32;
    v8f zero = {};
    v8f acc[2][2] = {{zero, zero}, {zero, zero}};
    gemm_tile_32x32(A, K, Wt, K, row0, col0, K, lane, acc);
    int n = lane & 15, mh = (lane >> 4) & 1;
    #pragma unroll
    for (int j = 0; j < 2; ++j) {
        int col = col0 + j * 16 + n;
        float bv = bias[col];
        #pragma unroll
        for (int i = 0; i < 2; ++i) {
            int rb = row0 + i * 16 + mh * 8;
            #pragma unroll
            for (int v = 0; v < 8; ++v) {
                float val = acc[i][j][v] + bv;
                val = (act == 0) ? tanhf(val) : (1.f / (1.f + __expf(-val)));
                outp[(long)(rb + v) * 256 + col] = val;
            }
        }
    }
}

// ---- scan phase 1: per-chunk reduce of h = a*h + b (a = 1-dt, b = dt*ut) ----
__global__ __launch_bounds__(256)
void scan_phase1(const float* __restrict__ ut, const float* __restrict__ dt,
                 float* __restrict__ chA, float* __restrict__ chB,
                 int L, int S, int CL, int NC) {
    int s = threadIdx.x;
    int b = blockIdx.x;
    int c = blockIdx.y;
    long base = ((long)b * L + (long)c * CL) * S + s;
    float Aacc = 1.f, Bacc = 0.f;
    for (int t = 0; t < CL; ++t) {
        long idx = base + (long)t * S;
        float d = dt[idx];
        float a = 1.f - d;
        Aacc = a * Aacc;
        Bacc = a * Bacc + d * ut[idx];
    }
    int ch = b * S + s;
    chA[(long)ch * NC + c] = Aacc;
    chB[(long)ch * NC + c] = Bacc;
}

// ---- scan phase 2: exclusive prefix over chunks per channel ----
__global__ __launch_bounds__(256)
void scan_phase2(const float* __restrict__ chA, const float* __restrict__ chB,
                 float* __restrict__ carry, int NC) {
    int ch = blockIdx.x * blockDim.x + threadIdx.x;
    float h = 0.f;
    for (int c = 0; c < NC; ++c) {
        carry[(long)ch * NC + c] = h;
        h = chA[(long)ch * NC + c] * h + chB[(long)ch * NC + c];
    }
}

// ---- scan phase 3: replay with carry, fuse y = h * sigmoid(g), cast bf16 ----
__global__ __launch_bounds__(256)
void scan_phase3(const float* __restrict__ ut, const float* __restrict__ dt,
                 const bf16_t* __restrict__ g, const float* __restrict__ carry,
                 bf16_t* __restrict__ ybf, int L, int S, int CL, int NC) {
    int s = threadIdx.x;
    int b = blockIdx.x;
    int c = blockIdx.y;
    int ch = b * S + s;
    float h = carry[(long)ch * NC + c];
    long base = ((long)b * L + (long)c * CL) * S + s;
    for (int t = 0; t < CL; ++t) {
        long idx = base + (long)t * S;
        float d = dt[idx];
        h = (1.f - d) * h + d * ut[idx];
        float gv = bf2f(g[idx]);
        float sg = 1.f / (1.f + __expf(-gv));
        ybf[idx] = f2bf(h * sg);
    }
}

// ---- kernel: out = residual + ybf @ W_out + b_out (K=256, N=1024) ----
__global__ __launch_bounds__(128)
void gemm_out_kernel(const bf16_t* __restrict__ ybf, const bf16_t* __restrict__ wt_out,
                     const float* __restrict__ bias, const float* __restrict__ resid,
                     float* __restrict__ outp, int N) {
    const int K = 256;
    int lane = threadIdx.x & 31;
    int wave = threadIdx.x >> 5;
    int row0 = blockIdx.x * 64 + (wave >> 1) * 32;
    int col0 = blockIdx.y * 64 + (wave & 1) * 32;
    v8f zero = {};
    v8f acc[2][2] = {{zero, zero}, {zero, zero}};
    gemm_tile_32x32(ybf, K, wt_out, K, row0, col0, K, lane, acc);
    int n = lane & 15, mh = (lane >> 4) & 1;
    #pragma unroll
    for (int j = 0; j < 2; ++j) {
        int col = col0 + j * 16 + n;
        float bv = bias[col];
        #pragma unroll
        for (int i = 0; i < 2; ++i) {
            int rb = row0 + i * 16 + mh * 8;
            #pragma unroll
            for (int v = 0; v < 8; ++v) {
                long idx = (long)(rb + v) * N + col;
                outp[idx] = resid[idx] + acc[i][j][v] + bv;
            }
        }
    }
}

extern "C" void kernel_launch(void* const* d_in, const int* in_sizes, int n_in,
                              void* d_out, int out_size, void* d_ws, size_t ws_size,
                              hipStream_t stream) {
    const int B = 8, L = 4096, D = 1024, S = 256;
    const int T = B * L;             // 32768 tokens
    const int CL = 256, NC = L / CL; // scan chunking: 16 chunks of 256

    const float* x      = (const float*)d_in[0];
    const float* w_norm = (const float*)d_in[1];
    const float* W_in   = (const float*)d_in[2];
    const float* b_in   = (const float*)d_in[3];
    const float* W_x    = (const float*)d_in[4];
    const float* b_x    = (const float*)d_in[5];
    const float* W_dt   = (const float*)d_in[6];
    const float* b_dt   = (const float*)d_in[7];
    const float* W_out  = (const float*)d_in[8];
    const float* b_out  = (const float*)d_in[9];
    float* out = (float*)d_out;

    // Workspace carve (256B aligned)
    char* p = (char*)d_ws;
    auto carve = [&](size_t n) { char* r = p; p += (n + 255) & ~(size_t)255; return (void*)r; };
    bf16_t* wt_in  = (bf16_t*)carve((size_t)3 * S * D * 2);  // 768 x 1024
    bf16_t* wt_x   = (bf16_t*)carve((size_t)S * S * 2);      // 256 x 256
    bf16_t* wt_dt  = (bf16_t*)carve((size_t)S * S * 2);      // 256 x 256
    bf16_t* wt_out = (bf16_t*)carve((size_t)D * S * 2);      // 1024 x 256
    bf16_t* xn     = (bf16_t*)carve((size_t)T * D * 2);
    bf16_t* u_bf   = (bf16_t*)carve((size_t)T * S * 2);
    bf16_t* z_bf   = (bf16_t*)carve((size_t)T * S * 2);
    bf16_t* g_bf   = (bf16_t*)carve((size_t)T * S * 2);
    float*  ut     = (float*)carve((size_t)T * S * 4);
    float*  dtb    = (float*)carve((size_t)T * S * 4);
    bf16_t* ybf    = (bf16_t*)carve((size_t)T * S * 2);
    float*  chA    = (float*)carve((size_t)B * S * NC * 4);
    float*  chB    = (float*)carve((size_t)B * S * NC * 4);
    float*  carry  = (float*)carve((size_t)B * S * NC * 4);

    // 1) Weight transpose+cast to bf16 (N x K layouts for WMMA B fragments)
    {
        long n1 = (long)D * 3 * S;
        convert_transpose_kernel<<<(n1 + 255) / 256, 256, 0, stream>>>(W_in, wt_in, D, 3 * S);
        long n2 = (long)S * S;
        convert_transpose_kernel<<<(n2 + 255) / 256, 256, 0, stream>>>(W_x, wt_x, S, S);
        convert_transpose_kernel<<<(n2 + 255) / 256, 256, 0, stream>>>(W_dt, wt_dt, S, S);
        long n3 = (long)S * D;
        convert_transpose_kernel<<<(n3 + 255) / 256, 256, 0, stream>>>(W_out, wt_out, S, D);
    }

    // 2) RMSNorm + bf16 cast
    rmsnorm_cast_kernel<<<T, 256, 0, stream>>>(x, w_norm, xn, D);

    // 3) uzg GEMM with TDM-staged weights: (32768 x 1024) @ (1024 x 768)
    gemm_in_kernel<<<dim3(T / 64, (3 * S) / 64), 128, 0, stream>>>(xn, wt_in, b_in, u_bf, z_bf, g_bf, D);

    // 4) Gate GEMMs with fused activations
    gemm_gate_kernel<<<dim3(T / 64, S / 64), 128, 0, stream>>>(u_bf, wt_x,  b_x,  ut,  0); // tanh
    gemm_gate_kernel<<<dim3(T / 64, S / 64), 128, 0, stream>>>(z_bf, wt_dt, b_dt, dtb, 1); // sigmoid

    // 5) Parallel scan over L (chunked linear recurrence), fused gate multiply
    scan_phase1<<<dim3(B, NC), S, 0, stream>>>(ut, dtb, chA, chB, L, S, CL, NC);
    scan_phase2<<<(B * S) / 256, 256, 0, stream>>>(chA, chB, carry, NC);
    scan_phase3<<<dim3(B, NC), S, 0, stream>>>(ut, dtb, g_bf, carry, ybf, L, S, CL, NC);

    // 6) Output GEMM + bias + residual
    gemm_out_kernel<<<dim3(T / 64, D / 64), 128, 0, stream>>>(ybf, wt_out, b_out, x, out, D);
}
